// PriorEncoder_26121991094738
// MI455X (gfx1250) — compile-verified, matrix operand order checked by reading, not run on hardware
//
#include <hip/hip_runtime.h>
#include <hip/hip_bf16.h>

// ---------------------------------------------------------------------------
// Problem constants (match the reference)
// ---------------------------------------------------------------------------
#define B_      16
#define TZ_     2048
#define TT_     512
#define D_      256
#define DFF_    1024
#define L_      6
#define HEADS_  4
#define DH_     64            // D/HEADS
#define CH_     256
#define HID_    256
#define KK_     5
#define FLOWS_  4
#define G_      256
#define NEGF    (-1000000000.0f)

typedef __attribute__((ext_vector_type(16))) _Float16 v16h;
typedef __attribute__((ext_vector_type(8)))  _Float16 v8h;
typedef __attribute__((ext_vector_type(8)))  float    v8f;

// ---------------------------------------------------------------------------
// Generic batched WMMA GEMM:  C[z] = act( scale * A[z] @ B[z] + bias )  (or +=)
//   A: row-major [M,K] with leading dim lda
//   B: transB==0 -> row-major [K,N] (ldb); transB==1 -> row-major [N,K] (ldb)
//   batch z decomposes: outer = z/innerCnt, inner = z%innerCnt
//   rows outside [m_lo, m_hi) contribute zeros (conv SAME padding)
// REQUIREMENTS (all call sites satisfy them): K % 32 == 0, N % 16 == 0.
// Block: 256 threads = 8 wave32 waves; 128x128 tile, BK=32.
// Wave w owns a 32x64 region: 2 A-frags x 4 B-frags -> 8 WMMAs / K-step.
// ---------------------------------------------------------------------------
#define BM 128
#define BN 128
#define BK 32
#define LSTR 40   // LDS row stride in halves (80B = 16B-aligned, bank-spread)

__device__ __forceinline__ v16h load_frag(const _Float16* rowbase, int hi) {
  // CDNA5 16-bit operand K-map: k = (e>=8)*16 + hi*8 + (e&7)
  v8h lo = *(const v8h*)(rowbase + hi * 8);
  v8h hb = *(const v8h*)(rowbase + 16 + hi * 8);
  return __builtin_shufflevector(lo, hb, 0,1,2,3,4,5,6,7,8,9,10,11,12,13,14,15);
}

__device__ __forceinline__ v8h cvt8(float4 f0, float4 f1) {
  v8h h;
  h[0] = (_Float16)f0.x; h[1] = (_Float16)f0.y;
  h[2] = (_Float16)f0.z; h[3] = (_Float16)f0.w;
  h[4] = (_Float16)f1.x; h[5] = (_Float16)f1.y;
  h[6] = (_Float16)f1.z; h[7] = (_Float16)f1.w;
  return h;
}

__global__ __launch_bounds__(256)
void gemm_wmma_kernel(const float* __restrict__ A, int lda, long long sAo, long long sAi,
                      const float* __restrict__ Bm, int ldb, long long sBo, long long sBi, int transB,
                      float* __restrict__ C, int ldc, long long sCo, long long sCi,
                      const float* __restrict__ bias,
                      int M, int N, int K, int innerCnt,
                      float scale, int accum, int act, int m_lo, int m_hi)
{
  __shared__ _Float16 As[BM * LSTR];
  __shared__ _Float16 Bs[BN * LSTR];

  const int z = blockIdx.z;
  const int outer = z / innerCnt;
  const int inner = z - outer * innerCnt;
  A  += (long long)outer * sAo + (long long)inner * sAi;
  Bm += (long long)outer * sBo + (long long)inner * sBi;
  C  += (long long)outer * sCo + (long long)inner * sCi;

  const int m0 = blockIdx.y * BM;
  const int n0 = blockIdx.x * BN;

  const int tid  = threadIdx.x;
  const int wave = tid >> 5;
  const int lane = tid & 31;
  const int lo16 = lane & 15;
  const int hi   = lane >> 4;

  const int wm = wave >> 1;        // 0..3 : M offset wm*32
  const int wn = wave & 1;         // 0..1 : N offset wn*64

  // A-stage addressing (row-guard only; K%32==0 so no K tail)
  const int arow = tid >> 1;              // 0..127
  const int acol = (tid & 1) * 16;        // 0 or 16
  const int agm  = m0 + arow;
  const bool arok = (agm < M) && (agm >= m_lo) && (agm < m_hi);
  const float* aP = A + (long long)agm * lda + acol;
  _Float16* aD = &As[arow * LSTR + acol];

  // B-stage addressing (row/chunk guards only)
  // transB: stage 128 rows (n) x 32 (k): row = tid>>1, col = (tid&1)*16
  const int btr_row = tid >> 1;           // n 0..127
  const int btr_col = (tid & 1) * 16;     // k 0 or 16
  const bool btr_ok = (n0 + btr_row) < N;
  const float* btrP = Bm + (long long)(n0 + btr_row) * ldb + btr_col;
  _Float16* btrD = &Bs[btr_row * LSTR + btr_col];

  // non-trans: stage 32 (k) x 128 (n): kk = tid>>3, nn = (tid&7)*16
  const int bk = tid >> 3;                // k 0..31
  const int bn = (tid & 7) * 16;          // n 0..112
  const bool bnt_ok = (n0 + bn) < N;      // chunk-level (N%16==0)
  const float* bP = Bm + (long long)bk * ldb + n0 + bn;

  v8f acc[2][4];
  #pragma unroll
  for (int i = 0; i < 2; ++i)
    #pragma unroll
    for (int j = 0; j < 4; ++j)
      acc[i][j] = (v8f){};

  for (int k0 = 0; k0 < K; k0 += BK) {
    // ---- stage A tile (128x32) -> LDS f16 ----------------------------------
    if (arok) {
      const float* ap = aP + k0;
      float4 f0 = *(const float4*)(ap);
      float4 f1 = *(const float4*)(ap + 4);
      float4 f2 = *(const float4*)(ap + 8);
      float4 f3 = *(const float4*)(ap + 12);
      *(v8h*)(aD)     = cvt8(f0, f1);
      *(v8h*)(aD + 8) = cvt8(f2, f3);
      if (k0 + BK < K) __builtin_prefetch(ap + BK, 0, 1);
    } else {
      v8h zz = {};
      *(v8h*)(aD)     = zz;
      *(v8h*)(aD + 8) = zz;
    }
    // ---- stage B tile -> LDS as Bs[n][k] -----------------------------------
    if (transB) {
      if (btr_ok) {
        const float* bp = btrP + k0;
        float4 f0 = *(const float4*)(bp);
        float4 f1 = *(const float4*)(bp + 4);
        float4 f2 = *(const float4*)(bp + 8);
        float4 f3 = *(const float4*)(bp + 12);
        *(v8h*)(btrD)     = cvt8(f0, f1);
        *(v8h*)(btrD + 8) = cvt8(f2, f3);
        if (k0 + BK < K) __builtin_prefetch(bp + BK, 0, 1);
      } else {
        v8h zz = {};
        *(v8h*)(btrD)     = zz;
        *(v8h*)(btrD + 8) = zz;
      }
    } else {
      if (bnt_ok) {
        const float* bp = bP + (long long)k0 * ldb;
        float4 f0 = *(const float4*)(bp);
        float4 f1 = *(const float4*)(bp + 4);
        float4 f2 = *(const float4*)(bp + 8);
        float4 f3 = *(const float4*)(bp + 12);
        Bs[(bn +  0) * LSTR + bk] = (_Float16)f0.x;
        Bs[(bn +  1) * LSTR + bk] = (_Float16)f0.y;
        Bs[(bn +  2) * LSTR + bk] = (_Float16)f0.z;
        Bs[(bn +  3) * LSTR + bk] = (_Float16)f0.w;
        Bs[(bn +  4) * LSTR + bk] = (_Float16)f1.x;
        Bs[(bn +  5) * LSTR + bk] = (_Float16)f1.y;
        Bs[(bn +  6) * LSTR + bk] = (_Float16)f1.z;
        Bs[(bn +  7) * LSTR + bk] = (_Float16)f1.w;
        Bs[(bn +  8) * LSTR + bk] = (_Float16)f2.x;
        Bs[(bn +  9) * LSTR + bk] = (_Float16)f2.y;
        Bs[(bn + 10) * LSTR + bk] = (_Float16)f2.z;
        Bs[(bn + 11) * LSTR + bk] = (_Float16)f2.w;
        Bs[(bn + 12) * LSTR + bk] = (_Float16)f3.x;
        Bs[(bn + 13) * LSTR + bk] = (_Float16)f3.y;
        Bs[(bn + 14) * LSTR + bk] = (_Float16)f3.z;
        Bs[(bn + 15) * LSTR + bk] = (_Float16)f3.w;
        if (k0 + BK < K) __builtin_prefetch(bp + (long long)BK * ldb, 0, 1);
      } else {
        const _Float16 zz = (_Float16)0.0f;
        #pragma unroll
        for (int i = 0; i < 16; ++i) Bs[(bn + i) * LSTR + bk] = zz;
      }
    }
    __syncthreads();

    // ---- fragments + 8 WMMAs (2 A-frags x 4 B-frags reuse) ------------------
    v16h a0 = load_frag(&As[(wm * 32 +      lo16) * LSTR], hi);
    v16h a1 = load_frag(&As[(wm * 32 + 16 + lo16) * LSTR], hi);
    v16h b0 = load_frag(&Bs[(wn * 64 +      lo16) * LSTR], hi);
    v16h b1 = load_frag(&Bs[(wn * 64 + 16 + lo16) * LSTR], hi);
    v16h b2 = load_frag(&Bs[(wn * 64 + 32 + lo16) * LSTR], hi);
    v16h b3 = load_frag(&Bs[(wn * 64 + 48 + lo16) * LSTR], hi);

    acc[0][0] = __builtin_amdgcn_wmma_f32_16x16x32_f16(false, a0, false, b0, (short)0, acc[0][0], false, false);
    acc[0][1] = __builtin_amdgcn_wmma_f32_16x16x32_f16(false, a0, false, b1, (short)0, acc[0][1], false, false);
    acc[0][2] = __builtin_amdgcn_wmma_f32_16x16x32_f16(false, a0, false, b2, (short)0, acc[0][2], false, false);
    acc[0][3] = __builtin_amdgcn_wmma_f32_16x16x32_f16(false, a0, false, b3, (short)0, acc[0][3], false, false);
    acc[1][0] = __builtin_amdgcn_wmma_f32_16x16x32_f16(false, a1, false, b0, (short)0, acc[1][0], false, false);
    acc[1][1] = __builtin_amdgcn_wmma_f32_16x16x32_f16(false, a1, false, b1, (short)0, acc[1][1], false, false);
    acc[1][2] = __builtin_amdgcn_wmma_f32_16x16x32_f16(false, a1, false, b2, (short)0, acc[1][2], false, false);
    acc[1][3] = __builtin_amdgcn_wmma_f32_16x16x32_f16(false, a1, false, b3, (short)0, acc[1][3], false, false);
    __syncthreads();
  }

  // ---- epilogue -------------------------------------------------------------
  #pragma unroll
  for (int i = 0; i < 2; ++i) {
    #pragma unroll
    for (int j = 0; j < 4; ++j) {
      const int gn = n0 + wn * 64 + j * 16 + lo16;
      if (gn < N) {
        const float bval = bias ? bias[gn] : 0.0f;
        #pragma unroll
        for (int r = 0; r < 8; ++r) {
          const int gm = m0 + wm * 32 + i * 16 + hi * 8 + r;  // C/D: m = hi*8 + vgpr
          if (gm < M) {
            const long long idx = (long long)gm * ldc + gn;
            float v = acc[i][j][r] * scale;
            if (accum) {
              v += C[idx];
            } else {
              v += bval;
              if (act == 1) v = fmaxf(v, 0.0f);
            }
            C[idx] = v;
          }
        }
      }
    }
  }
}

// ---------------------------------------------------------------------------
// Small helper kernels
// ---------------------------------------------------------------------------
__device__ __forceinline__ float block_reduce_sum(float v, float* sm) {
  int tid = threadIdx.x;
  sm[tid] = v; __syncthreads();
  for (int s = blockDim.x >> 1; s > 0; s >>= 1) {
    if (tid < s) sm[tid] += sm[tid + s];
    __syncthreads();
  }
  float r = sm[0]; __syncthreads();
  return r;
}
__device__ __forceinline__ float block_reduce_max(float v, float* sm) {
  int tid = threadIdx.x;
  sm[tid] = v; __syncthreads();
  for (int s = blockDim.x >> 1; s > 0; s >>= 1) {
    if (tid < s) sm[tid] = fmaxf(sm[tid], sm[tid + s]);
    __syncthreads();
  }
  float r = sm[0]; __syncthreads();
  return r;
}

// h[b,t,:] = embed[text]*sqrt(D) + pos_encoding(t)
__global__ void embed_pos_kernel(const int* __restrict__ text,
                                 const float* __restrict__ embed,
                                 float* __restrict__ h) {
  const int row = blockIdx.x;          // b*TT + t
  const int t = row % TT_;
  const int d = threadIdx.x;
  const int tok = text[row];
  const float ang = (float)t * powf(10000.0f, -(float)(2 * (d / 2)) / (float)D_);
  const float pe = (d & 1) ? cosf(ang) : sinf(ang);
  h[(long long)row * D_ + d] = embed[(long long)tok * D_ + d] * 16.0f + pe;
}

// h = LN(h + res) * g + b   (one block per row, 256 threads == D)
__global__ void ln_res_kernel(float* __restrict__ h, const float* __restrict__ res,
                              const float* __restrict__ g, const float* __restrict__ b) {
  __shared__ float sm[256];
  const long long row = blockIdx.x;
  const int d = threadIdx.x;
  float x = h[row * D_ + d] + res[row * D_ + d];
  const float mean = block_reduce_sum(x, sm) * (1.0f / (float)D_);
  const float dv = x - mean;
  const float var = block_reduce_sum(dv * dv, sm) * (1.0f / (float)D_);
  h[row * D_ + d] = g[d] * dv * rsqrtf(var + 1e-6f) + b[d];
}

// masked softmax over TT keys; one block (256 thr) per (b,h,q) row
__global__ void softmax_mask_kernel(float* __restrict__ scores,
                                    const int* __restrict__ text) {
  __shared__ float sm[256];
  const long long row = blockIdx.x;                 // over B*HEADS*TT
  const int b = (int)(row / (HEADS_ * TT_));
  float* p = scores + row * TT_;
  const int* tm = text + (long long)b * TT_;
  const int tid = threadIdx.x;
  float v0 = p[tid]       + (tm[tid]       != 0 ? 0.0f : NEGF);
  float v1 = p[tid + 256] + (tm[tid + 256] != 0 ? 0.0f : NEGF);
  const float mx = block_reduce_max(fmaxf(v0, v1), sm);
  const float e0 = __expf(v0 - mx);
  const float e1 = __expf(v1 - mx);
  const float s = block_reduce_sum(e0 + e1, sm);
  const float inv = 1.0f / s;
  p[tid] = e0 * inv;
  p[tid + 256] = e1 * inv;
}

__global__ void copy_kernel(const float* __restrict__ src, float* __restrict__ dst, long long n) {
  long long i = (long long)blockIdx.x * blockDim.x + threadIdx.x;
  if (i < n) dst[i] = src[i];
}

// hh[b,t,:] += ccond[b,:]
__global__ void add_cond_kernel(float* __restrict__ hh, const float* __restrict__ cc) {
  const long long row = blockIdx.x;            // b*TZ + t
  const int b = (int)(row / TZ_);
  const int d = threadIdx.x;
  hh[row * HID_ + d] += cc[(long long)b * HID_ + d];
}

__global__ void tanh_kernel(float* __restrict__ x, long long n) {
  long long i = (long long)blockIdx.x * blockDim.x + threadIdx.x;
  if (i < n) x[i] = tanhf(x[i]);
}

// zc_new = concat([x1 + m, x0])
__global__ void zc_update_kernel(const float* __restrict__ cur,
                                 const float* __restrict__ m,
                                 float* __restrict__ nxt) {
  const long long row = blockIdx.x;
  const int c = threadIdx.x;
  const float* ci = cur + row * CH_;
  float* no = nxt + row * CH_;
  if (c < CH_ / 2) no[c] = ci[CH_ / 2 + c] + m[row * (CH_ / 2) + c];
  else             no[c] = ci[c - CH_ / 2];
}

// spr = exp(-logvar); msp = mu*spr
__global__ void spr_msp_kernel(const float* __restrict__ mu, const float* __restrict__ lv,
                               float* __restrict__ spr, float* __restrict__ msp, long long n) {
  long long i = (long long)blockIdx.x * blockDim.x + threadIdx.x;
  if (i < n) {
    float s = __expf(-lv[i]);
    spr[i] = s;
    msp[i] = mu[i] * s;
  }
}

__global__ void zcsq_kernel(const float* __restrict__ zc, float* __restrict__ zq, long long n) {
  long long i = (long long)blockIdx.x * blockDim.x + threadIdx.x;
  if (i < n) { float v = zc[i]; zq[i] = -0.5f * v * v; }
}

// vec1[b,j] = sum_d( -0.5*log(2pi) - 0.5*logvar - 0.5*mu^2*spr )
__global__ void vec1_kernel(const float* __restrict__ mu, const float* __restrict__ lv,
                            const float* __restrict__ spr, float* __restrict__ vec1) {
  __shared__ float sm[256];
  const long long row = blockIdx.x;            // b*TT + j
  const int d = threadIdx.x;
  const long long i = row * D_ + d;
  const float term = -0.91893853320467274f - 0.5f * lv[i] - 0.5f * mu[i] * mu[i] * spr[i];
  const float s = block_reduce_sum(term, sm);
  if (d == 0) vec1[row] = s;
}

// logp[b,t,j] += vec1[b,j]; mask
__global__ void finalize_logp_kernel(float* __restrict__ logp, const float* __restrict__ vec1,
                                     const int* __restrict__ text) {
  const long long row = blockIdx.x;            // b*TZ + t
  const int b = (int)(row / TZ_);
  float* p = logp + row * TT_;
  for (int j = threadIdx.x; j < TT_; j += 256) {
    float v = p[j] + vec1[(long long)b * TT_ + j];
    p[j] = (text[(long long)b * TT_ + j] != 0) ? v : NEGF;
  }
}

// MAS forward DP: one block (TT threads) per batch, sequential in t with LDS Q
__global__ void mas_fwd_kernel(const float* __restrict__ logp,
                               unsigned char* __restrict__ choices) {
  __shared__ float Q[TT_];
  const int b = blockIdx.x;
  const int j = threadIdx.x;
  const float* lp = logp + (long long)b * TZ_ * TT_;
  Q[j] = (j == 0) ? lp[0] : NEGF;
  __syncthreads();
  unsigned char* ch = choices + (long long)b * (TZ_ - 1) * TT_;
  for (int t = 1; t < TZ_; ++t) {
    const float qj = Q[j];
    const float qm = (j > 0) ? Q[j - 1] : NEGF;
    __syncthreads();
    ch[(long long)(t - 1) * TT_ + j] = (qm > qj) ? 1 : 0;
    Q[j] = lp[(long long)t * TT_ + j] + fmaxf(qj, qm);
    __syncthreads();
  }
}

// MAS backtrack: one thread per batch
__global__ void mas_bwd_kernel(const int* __restrict__ text,
                               const unsigned char* __restrict__ choices,
                               int* __restrict__ path) {
  const int b = threadIdx.x;
  if (b >= B_) return;
  int cnt = 0;
  for (int j = 0; j < TT_; ++j) cnt += (text[(long long)b * TT_ + j] != 0);
  int j = cnt - 1;
  path[(long long)b * TZ_ + TZ_ - 1] = j;
  const unsigned char* ch = choices + (long long)b * (TZ_ - 1) * TT_;
  for (int t = TZ_ - 1; t >= 1; --t) {
    j -= (int)ch[(long long)(t - 1) * TT_ + j];
    path[(long long)b * TZ_ + t - 1] = j;
  }
}

__global__ void gather_ctx_kernel(const float* __restrict__ mu, const int* __restrict__ path,
                                  float* __restrict__ ctx) {
  const long long row = blockIdx.x;            // b*TZ + t
  const int b = (int)(row / TZ_);
  const int j = path[row];
  ctx[row * D_ + threadIdx.x] = mu[((long long)b * TT_ + j) * D_ + threadIdx.x];
}

// ---------------------------------------------------------------------------
// Host side
// ---------------------------------------------------------------------------
static void launch_gemm(hipStream_t s,
                        const float* A, int lda, long long sAo, long long sAi,
                        const float* Bm, int ldb, long long sBo, long long sBi, int transB,
                        float* C, int ldc, long long sCo, long long sCi,
                        const float* bias, int M, int N, int K,
                        int batch, int innerCnt,
                        float scale, int accum, int act, int m_lo, int m_hi)
{
  dim3 grid((N + BN - 1) / BN, (M + BM - 1) / BM, batch);
  gemm_wmma_kernel<<<grid, dim3(256), 0, s>>>(A, lda, sAo, sAi, Bm, ldb, sBo, sBi, transB,
                                              C, ldc, sCo, sCi, bias, M, N, K, innerCnt,
                                              scale, accum, act, m_lo, m_hi);
}

extern "C" void kernel_launch(void* const* d_in, const int* in_sizes, int n_in,
                              void* d_out, int out_size, void* d_ws, size_t ws_size,
                              hipStream_t stream) {
  (void)in_sizes; (void)n_in; (void)out_size; (void)ws_size;

  const float* z        = (const float*)d_in[0];
  const int*   text     = (const int*)  d_in[1];
  const float* cond     = (const float*)d_in[2];
  const float* embed    = (const float*)d_in[3];
  const float* wq       = (const float*)d_in[4];
  const float* wk       = (const float*)d_in[5];
  const float* wv       = (const float*)d_in[6];
  const float* wo       = (const float*)d_in[7];
  const float* bq       = (const float*)d_in[8];
  const float* bk       = (const float*)d_in[9];
  const float* bv       = (const float*)d_in[10];
  const float* bo       = (const float*)d_in[11];
  const float* ln1g     = (const float*)d_in[12];
  const float* ln1b     = (const float*)d_in[13];
  const float* ln2g     = (const float*)d_in[14];
  const float* ln2b     = (const float*)d_in[15];
  const float* w1       = (const float*)d_in[16];
  const float* b1       = (const float*)d_in[17];
  const float* w2       = (const float*)d_in[18];
  const float* b2       = (const float*)d_in[19];
  const float* mu_w     = (const float*)d_in[20];
  const float* mu_b     = (const float*)d_in[21];
  const float* lv_w     = (const float*)d_in[22];
  const float* lv_b     = (const float*)d_in[23];
  const float* flow_in  = (const float*)d_in[24];
  const float* flow_inb = (const float*)d_in[25];
  const float* flow_cd  = (const float*)d_in[26];
  const float* flow_cv  = (const float*)d_in[27];
  const float* flow_cvb = (const float*)d_in[28];
  const float* flow_out = (const float*)d_in[29];
  const float* flow_ob  = (const float*)d_in[30];

  float* ws = (float*)d_ws;

  // Output layout: [ctx (B*TZ*D), h (B*TT*D)]
  const long long CTX_N = (long long)B_ * TZ_ * D_;      // 8,388,608
  float* ctx  = (float*)d_out;
  float* hout = (float*)d_out + CTX_N;

  // Workspace layout (float offsets)
  const long long MU_OFF = 0;                    // 2,097,152
  const long long LV_OFF = 2097152;              // 2,097,152
  const long long QB_OFF = 4194304;              // 2,097,152
  const long long KB_OFF = 6291456;              // 2,097,152  (later: spr)
  const long long VB_OFF = 8388608;              // 2,097,152  (later: msp)
  const long long SC_OFF = 10485760;             // 16,777,216 (scores -> flows scratch -> logp)
  const long long AT_OFF = 27262976;             // 2,097,152
  const long long FF_OFF = 29360128;             // 8,388,608  (ffn hidden -> hh -> MAS choices)
  const long long Z0_OFF = 37748736;             // 8,388,608
  const long long Z1_OFF = 46137344;             // 8,388,608  (zc pong -> zcsq)
  const long long V1_OFF = 54525952;             // 8,192
  const long long PT_OFF = 54534144;             // 32,768 ints

  float* MU = ws + MU_OFF;
  float* LV = ws + LV_OFF;
  float* QB = ws + QB_OFF;
  float* KB = ws + KB_OFF;
  float* VB = ws + VB_OFF;
  float* SC = ws + SC_OFF;
  float* AT = ws + AT_OFF;
  float* FF = ws + FF_OFF;
  float* Z0 = ws + Z0_OFF;
  float* Z1 = ws + Z1_OFF;
  float* V1 = ws + V1_OFF;
  int*   PT = (int*)(ws + PT_OFF);
  // flows sub-allocations inside SC (free until logp phase)
  float* H2 = SC;                 // conv output [B,TZ,HID]
  float* MB = SC + 8388608;       // m [B,TZ,CH/2]
  float* CC = SC + 12582912;      // cond@flow_cond [B,HID]
  unsigned char* CHO = (unsigned char*)FF;

  const int MTOK = B_ * TT_;      // 8192 token rows
  const int MROW = B_ * TZ_;      // 32768 z rows

  // ---- 1. embedding + positional encoding -> h -----------------------------
  embed_pos_kernel<<<MTOK, D_, 0, stream>>>(text, embed, hout);

  // ---- 2. transformer encoder ----------------------------------------------
  for (int l = 0; l < L_; ++l) {
    const float* wql = wq + (long long)l * D_ * D_;
    const float* wkl = wk + (long long)l * D_ * D_;
    const float* wvl = wv + (long long)l * D_ * D_;
    const float* wol = wo + (long long)l * D_ * D_;

    launch_gemm(stream, hout, D_, 0, 0, wql, D_, 0, 0, 0, QB, D_, 0, 0,
                bq + (long long)l * D_, MTOK, D_, D_, 1, 1, 1.0f, 0, 0, 0, MTOK);
    launch_gemm(stream, hout, D_, 0, 0, wkl, D_, 0, 0, 0, KB, D_, 0, 0,
                bk + (long long)l * D_, MTOK, D_, D_, 1, 1, 1.0f, 0, 0, 0, MTOK);
    launch_gemm(stream, hout, D_, 0, 0, wvl, D_, 0, 0, 0, VB, D_, 0, 0,
                bv + (long long)l * D_, MTOK, D_, D_, 1, 1, 1.0f, 0, 0, 0, MTOK);

    // scores[b,h] = (q @ k^T) / sqrt(dh)
    launch_gemm(stream,
                QB, D_, (long long)TT_ * D_, DH_,
                KB, D_, (long long)TT_ * D_, DH_, 1,
                SC, TT_, (long long)HEADS_ * TT_ * TT_, (long long)TT_ * TT_,
                nullptr, TT_, TT_, DH_, B_ * HEADS_, HEADS_,
                0.125f, 0, 0, 0, TT_);

    softmax_mask_kernel<<<B_ * HEADS_ * TT_, 256, 0, stream>>>(SC, text);

    // att[b,t,h*dh] = attn @ v
    launch_gemm(stream,
                SC, TT_, (long long)HEADS_ * TT_ * TT_, (long long)TT_ * TT_,
                VB, D_, (long long)TT_ * D_, DH_, 0,
                AT, D_, (long long)TT_ * D_, DH_,
                nullptr, TT_, DH_, TT_, B_ * HEADS_, HEADS_,
                1.0f, 0, 0, 0, TT_);

    // output projection
    launch_gemm(stream, AT, D_, 0, 0, wol, D_, 0, 0, 0, QB, D_, 0, 0,
                bo + (long long)l * D_, MTOK, D_, D_, 1, 1, 1.0f, 0, 0, 0, MTOK);

    ln_res_kernel<<<MTOK, D_, 0, stream>>>(hout, QB, ln1g + (long long)l * D_,
                                           ln1b + (long long)l * D_);

    // FFN
    launch_gemm(stream, hout, D_, 0, 0, w1 + (long long)l * D_ * DFF_, DFF_, 0, 0, 0,
                FF, DFF_, 0, 0, b1 + (long long)l * DFF_,
                MTOK, DFF_, D_, 1, 1, 1.0f, 0, 1, 0, MTOK);
    launch_gemm(stream, FF, DFF_, 0, 0, w2 + (long long)l * DFF_ * D_, D_, 0, 0, 0,
                QB, D_, 0, 0, b2 + (long long)l * D_,
                MTOK, D_, DFF_, 1, 1, 1.0f, 0, 0, 0, MTOK);

    ln_res_kernel<<<MTOK, D_, 0, stream>>>(hout, QB, ln2g + (long long)l * D_,
                                           ln2b + (long long)l * D_);
  }

  // ---- 3. mu / logvar -------------------------------------------------------
  launch_gemm(stream, hout, D_, 0, 0, mu_w, D_, 0, 0, 0, MU, D_, 0, 0,
              mu_b, MTOK, D_, D_, 1, 1, 1.0f, 0, 0, 0, MTOK);
  launch_gemm(stream, hout, D_, 0, 0, lv_w, D_, 0, 0, 0, LV, D_, 0, 0,
              lv_b, MTOK, D_, D_, 1, 1, 1.0f, 0, 0, 0, MTOK);

  // ---- 4. flows -------------------------------------------------------------
  {
    const long long nZ = (long long)B_ * TZ_ * CH_;
    copy_kernel<<<(int)((nZ + 255) / 256), 256, 0, stream>>>(z, Z0, nZ);
  }
  float* cur = Z0;
  float* nxt = Z1;
  for (int f = 0; f < FLOWS_; ++f) {
    // ccond = cond @ flow_cond[f]
    launch_gemm(stream, cond, G_, 0, 0, flow_cd + (long long)f * G_ * HID_, HID_, 0, 0, 0,
                CC, HID_, 0, 0, nullptr, B_, HID_, G_, 1, 1, 1.0f, 0, 0, 0, B_);
    // hh = x0 @ flow_in[f] + b
    launch_gemm(stream, cur, CH_, 0, 0, flow_in + (long long)f * (CH_ / 2) * HID_, HID_, 0, 0, 0,
                FF, HID_, 0, 0, flow_inb + (long long)f * HID_,
                MROW, HID_, CH_ / 2, 1, 1, 1.0f, 0, 0, 0, MROW);
    add_cond_kernel<<<MROW, HID_, 0, stream>>>(FF, CC);
    // SAME conv over time as 5 accumulated GEMMs with row-validity windows
    for (int k = 0; k < KK_; ++k) {
      const int sh = k - 2;
      const int mlo = (sh < 0) ? -sh : 0;
      const int mhi = (sh > 0) ? (TZ_ - sh) : TZ_;
      launch_gemm(stream,
                  FF + (long long)sh * HID_, HID_, (long long)TZ_ * HID_, 0,
                  flow_cv + ((long long)f * KK_ + k) * HID_ * HID_, HID_, 0, 0, 0,
                  H2, HID_, (long long)TZ_ * HID_, 0,
                  (k == 0) ? (flow_cvb + (long long)f * HID_) : nullptr,
                  TZ_, HID_, HID_, B_, 1,
                  1.0f, (k == 0) ? 0 : 1, 0, mlo, mhi);
    }
    {
      const long long n = (long long)B_ * TZ_ * HID_;
      tanh_kernel<<<(int)((n + 255) / 256), 256, 0, stream>>>(H2, n);
    }
    // m = tanh(...) @ flow_out[f] + b
    launch_gemm(stream, H2, HID_, 0, 0, flow_out + (long long)f * HID_ * (CH_ / 2), CH_ / 2, 0, 0, 0,
                MB, CH_ / 2, 0, 0, flow_ob + (long long)f * (CH_ / 2),
                MROW, CH_ / 2, HID_, 1, 1, 1.0f, 0, 0, 0, MROW);
    zc_update_kernel<<<MROW, CH_, 0, stream>>>(cur, MB, nxt);
    float* t = cur; cur = nxt; nxt = t;
  }
  // after 4 flows cur == Z0 (even number of swaps)

  // ---- 5. log-likelihood logp[b,t,j] ---------------------------------------
  {
    const long long n = (long long)B_ * TT_ * D_;
    spr_msp_kernel<<<(int)((n + 255) / 256), 256, 0, stream>>>(MU, LV, KB, VB, n);
  }
  {
    const long long n = (long long)B_ * TZ_ * CH_;
    zcsq_kernel<<<(int)((n + 255) / 256), 256, 0, stream>>>(cur, Z1, n);
  }
  vec1_kernel<<<MTOK, D_, 0, stream>>>(MU, LV, KB, V1);

  // nc2: logp = (-0.5 zc^2) @ spr^T
  launch_gemm(stream,
              Z1, CH_, (long long)TZ_ * CH_, 0,
              KB, D_, (long long)TT_ * D_, 0, 1,
              SC, TT_, (long long)TZ_ * TT_, 0,
              nullptr, TZ_, TT_, CH_, B_, 1, 1.0f, 0, 0, 0, TZ_);
  // nc3: logp += zc @ (mu*spr)^T
  launch_gemm(stream,
              cur, CH_, (long long)TZ_ * CH_, 0,
              VB, D_, (long long)TT_ * D_, 0, 1,
              SC, TT_, (long long)TZ_ * TT_, 0,
              nullptr, TZ_, TT_, CH_, B_, 1, 1.0f, 1, 0, 0, TZ_);

  finalize_logp_kernel<<<MROW, 256, 0, stream>>>(SC, V1, text);

  // ---- 6. monotonic alignment search + gather ------------------------------
  mas_fwd_kernel<<<B_, TT_, 0, stream>>>(SC, CHO);
  mas_bwd_kernel<<<1, 32, 0, stream>>>(text, CHO, PT);
  gather_ctx_kernel<<<MROW, D_, 0, stream>>>(MU, PT, ctx);
}
